// EarthSpecificBlock_21552145891642
// MI455X (gfx1250) — matrix operand order repaired and checked by reading, not compiled, not run on hardware
//
#include <hip/hip_runtime.h>
#include <hip/hip_bf16.h>

typedef __attribute__((ext_vector_type(16))) _Float16 v16h;
typedef __attribute__((ext_vector_type(8)))  float    v8f;
typedef __attribute__((ext_vector_type(4)))  int      v4i;

#define WMMA_F16(a, b, c) \
  __builtin_amdgcn_wmma_f32_16x16x32_f16(false, (a), false, (b), (short)0, (c), false, false)

// ---------------- CDNA5 async global->LDS staging (guarded) ----------------
#if __has_builtin(__builtin_amdgcn_global_load_async_to_lds_b128) && \
    __has_builtin(__builtin_amdgcn_s_wait_asynccnt)
#define HAVE_ASYNC_LDS 1
typedef __attribute__((address_space(1))) v4i gv4i;   // global int4
typedef __attribute__((address_space(3))) v4i lv4i;   // LDS int4
#endif

// copy one 16B chunk global->LDS
__device__ __forceinline__ void cp16_g2l(void* lds_dst, const void* gsrc) {
#ifdef HAVE_ASYNC_LDS
  __builtin_amdgcn_global_load_async_to_lds_b128((gv4i*)gsrc, (lv4i*)lds_dst, 0, 0);
#else
  *(uint4*)lds_dst = *(const uint4*)gsrc;
#endif
}
// wait for this wave's async copies, then workgroup barrier
__device__ __forceinline__ void stage_wait() {
#ifdef HAVE_ASYNC_LDS
  __builtin_amdgcn_s_wait_asynccnt(0);
#endif
  __syncthreads();
}

// ---------------- problem constants ----------------
#define CDIM   192
#define NHEAD  6
#define HD     32
#define LWIN   144      // 2*6*12 tokens per window
#define LPAD   160      // K-padded window length for P@V
#define NWIN   960      // 4 * 16 * 15
#define TOK    131040   // 8*91*180
#define CH     768      // 4*C
#define HDIM 91
#define WDIM 180
#define HP 96

// ---------------- WMMA fragment helpers (CDNA5 16x16x32 f16 layouts) ----------------
__device__ __forceinline__ v16h load_frag_a(const _Float16* A, int lda, int m0, int k0, int lane) {
  const _Float16* p = A + (long)(m0 + (lane & 15)) * lda + k0 + 16 * (lane >> 4);
  v16h r;
#pragma unroll
  for (int i = 0; i < 16; ++i) r[i] = p[i];
  return r;
}
__device__ __forceinline__ v16h load_frag_b(const _Float16* Bm, int ldb, int n0, int k0, int lane) {
  const _Float16* p = Bm + (long)(n0 + (lane & 15)) * ldb + k0 + 16 * (lane >> 4);
  v16h r;
#pragma unroll
  for (int i = 0; i < 16; ++i) r[i] = p[i];
  return r;
}
__device__ __forceinline__ void store_frag_f32(float* C, int ldc, int m0, int n0, int lane, v8f acc) {
  int n = n0 + (lane & 15);
  int mb = m0 + 8 * (lane >> 4);
#pragma unroll
  for (int j = 0; j < 8; ++j) C[(long)(mb + j) * ldc + n] = acc[j];
}

__device__ __forceinline__ float wave_sum(float v) {
#pragma unroll
  for (int off = 16; off > 0; off >>= 1) v += __shfl_xor(v, off, 32);
  return v;
}
__device__ __forceinline__ float wave_max(float v) {
#pragma unroll
  for (int off = 16; off > 0; off >>= 1) v = fmaxf(v, __shfl_xor(v, off, 32));
  return v;
}

// (window, token) -> source token in original x (roll -1,-3,-6 over padded 8x96x180)
__device__ __forceinline__ int win_src_token(int w, int t, bool& valid) {
  int wz = w / 240, wh = (w / 15) % 16, ww = w % 15;
  int tz = t / 72, th = (t / 12) % 6, tw = t % 12;
  int z = wz * 2 + tz, h = wh * 6 + th, x = ww * 12 + tw;
  int zs = (z + 1) & 7;
  int hs = h + 3; if (hs >= HP) hs -= HP;
  int xs = x + 6; if (xs >= WDIM) xs -= WDIM;
  valid = (hs < HDIM);
  return (zs * HDIM + hs) * WDIM + xs;
}
// original token -> (window, token) after inverse roll
__device__ __forceinline__ void tok_to_win(int n, int& w, int& t) {
  int xs = n % WDIM; int hs = (n / WDIM) % HDIM; int zs = n / (WDIM * HDIM);
  int z = (zs + 7) & 7;
  int h = hs - 3; if (h < 0) h += HP;
  int x = xs - 6; if (x < 0) x += WDIM;
  w = (z >> 1) * 240 + (h / 6) * 15 + x / 12;
  t = (z & 1) * 72 + (h % 6) * 12 + x % 12;
}

// ---------------- kernels ----------------
__global__ void cvt_f16_kernel(const float* __restrict__ src, _Float16* __restrict__ dst, int n) {
  int i = blockIdx.x * blockDim.x + threadIdx.x;
  if (i < n) dst[i] = (_Float16)src[i];
}

// fused window-gather + LayerNorm -> xw f16 [w][t][C]; one wave per token
__global__ void ln_window_kernel(const float* __restrict__ x, const float* __restrict__ g,
                                 const float* __restrict__ b, _Float16* __restrict__ xw) {
  int wv = (blockIdx.x * blockDim.x + threadIdx.x) >> 5;
  int lane = threadIdx.x & 31;
  if (wv >= NWIN * LWIN) return;
  int w = wv / LWIN, t = wv % LWIN;
  bool valid; int src = win_src_token(w, t, valid);
  const float* xp = x + (long)src * CDIM;
  float vals[6];
#pragma unroll
  for (int i = 0; i < 6; ++i) vals[i] = valid ? xp[lane + 32 * i] : 0.f;
  float s = 0.f;
#pragma unroll
  for (int i = 0; i < 6; ++i) s += vals[i];
  float mean = wave_sum(s) * (1.f / CDIM);
  float vs = 0.f;
#pragma unroll
  for (int i = 0; i < 6; ++i) { float d = vals[i] - mean; vs += d * d; }
  float inv = rsqrtf(wave_sum(vs) * (1.f / CDIM) + 1e-5f);
  _Float16* op = xw + ((long)w * LWIN + t) * CDIM;
#pragma unroll
  for (int i = 0; i < 6; ++i) {
    int c = lane + 32 * i;
    op[c] = (_Float16)((vals[i] - mean) * inv * g[c] + b[c]);
  }
}

// per-window QKV GEMM: (144x192) @ (192x576)^T-layout; scatter q(scaled)/k/v [w][h][t][d]
// register-blocked: B fragments hoisted per N-tile, reused across all 9 M-tiles
__global__ void qkv_kernel(const _Float16* __restrict__ xw, const _Float16* __restrict__ wq,
                           const float* __restrict__ bq, _Float16* __restrict__ q,
                           _Float16* __restrict__ k, _Float16* __restrict__ v) {
  extern __shared__ __align__(16) char smem[];
  _Float16* As = (_Float16*)smem;                       // 144x192
  int w = blockIdx.x, tid = threadIdx.x;
  const uint4* srcv = (const uint4*)(xw + (long)w * LWIN * CDIM);
  for (int i = tid; i < LWIN * CDIM / 8; i += 256) cp16_g2l((uint4*)As + i, srcv + i);
  stage_wait();
  int wave = tid >> 5, lane = tid & 31;
  const float scale = 0.17677669529663687f;             // 32^-0.5
  for (int nt = wave; nt < 36; nt += 8) {
    v16h bf[6];
#pragma unroll
    for (int kk = 0; kk < 6; ++kk) bf[kk] = load_frag_b(wq, CDIM, nt * 16, kk * 32, lane);
    int n = nt * 16 + (lane & 15);
    int which = n / CDIM, cc = n % CDIM, head = cc >> 5, d = cc & 31;
    float bias = bq[n];
    _Float16* dst = (which == 0) ? q : (which == 1) ? k : v;
    float sc = (which == 0) ? scale : 1.f;
    long base = ((long)w * NHEAD + head) * LWIN * HD + d;
    for (int mt = 0; mt < 9; ++mt) {
      v8f acc = {};
#pragma unroll
      for (int kk = 0; kk < 6; ++kk) {
        v16h a = load_frag_a(As, CDIM, mt * 16, kk * 32, lane);
        acc = WMMA_F16(a, bf[kk], acc);
      }
      int mb = mt * 16 + 8 * (lane >> 4);
#pragma unroll
      for (int j = 0; j < 8; ++j) dst[base + (long)(mb + j) * HD] = (_Float16)((acc[j] + bias) * sc);
    }
  }
}

// per-(window,head) attention. LDS: q(9216) k(9216) vT(10240) S f32(92160) P f16(46080)
__global__ void attn_kernel(const _Float16* __restrict__ q, const _Float16* __restrict__ k,
                            const _Float16* __restrict__ v, _Float16* __restrict__ attn_out) {
  extern __shared__ __align__(16) char smem[];
  _Float16* qs = (_Float16*)smem;
  _Float16* ks = qs + LWIN * HD;
  _Float16* vT = ks + LWIN * HD;                        // 32 x 160 (transposed, K-padded)
  float*    S  = (float*)(smem + 28672);                // 144 x 160
  _Float16* P  = (_Float16*)(smem + 120832);            // 144 x 160
  int wh = blockIdx.x, tid = threadIdx.x, wave = tid >> 5, lane = tid & 31;
  const _Float16* qg = q + (long)wh * LWIN * HD;
  const _Float16* kg = k + (long)wh * LWIN * HD;
  const _Float16* vg = v + (long)wh * LWIN * HD;
  for (int i = tid; i < LWIN * HD / 8; i += 256) {
    cp16_g2l((uint4*)qs + i, (const uint4*)qg + i);
    cp16_g2l((uint4*)ks + i, (const uint4*)kg + i);
  }
  for (int i = tid; i < HD * LPAD; i += 256) {          // transpose v, zero K-pad rows
    int d = i / LPAD, t = i % LPAD;
    vT[i] = (t < LWIN) ? vg[t * HD + d] : (_Float16)0.f;
  }
  stage_wait();
  // S = q @ k^T  (9x9 tiles, K=32)
  for (int tile = wave; tile < 81; tile += 8) {
    int mt = tile / 9, nt = tile % 9;
    v8f acc = {};
    v16h a  = load_frag_a(qs, HD, mt * 16, 0, lane);
    v16h bf = load_frag_b(ks, HD, nt * 16, 0, lane);
    acc = WMMA_F16(a, bf, acc);
    store_frag_f32(S, LPAD, mt * 16, nt * 16, lane, acc);
  }
  __syncthreads();
  // row softmax (f32), write P f16 with zero K-pad columns
  for (int row = wave; row < LWIN; row += 8) {
    float* sr = S + row * LPAD;
    _Float16* pr = P + row * LPAD;
    float m = -1e30f;
    for (int c = lane; c < LWIN; c += 32) m = fmaxf(m, sr[c]);
    m = wave_max(m);
    float sum = 0.f;
    for (int c = lane; c < LPAD; c += 32) {
      float e = (c < LWIN) ? __expf(sr[c] - m) : 0.f;
      sum += e;
      pr[c] = (_Float16)e;
    }
    float inv = 1.f / wave_sum(sum);
    for (int c = lane; c < LPAD; c += 32) pr[c] = (_Float16)((float)pr[c] * inv);
  }
  __syncthreads();
  // out = P @ v  (9x2 tiles, K=160)
  int wdx = wh / NHEAD, head = wh % NHEAD;
  for (int tile = wave; tile < 18; tile += 8) {
    int mt = tile / 2, nt = tile % 2;
    v8f acc = {};
#pragma unroll
    for (int kk = 0; kk < LPAD; kk += 32) {
      v16h a  = load_frag_a(P, LPAD, mt * 16, kk, lane);
      v16h bf = load_frag_b(vT, LPAD, nt * 16, kk, lane);
      acc = WMMA_F16(a, bf, acc);
    }
    int col = head * HD + nt * 16 + (lane & 15);
    int mb = mt * 16 + 8 * (lane >> 4);
    _Float16* op = attn_out + (long)wdx * LWIN * CDIM + col;
#pragma unroll
    for (int j = 0; j < 8; ++j) op[(long)(mb + j) * CDIM] = (_Float16)acc[j];
  }
}

// per-window proj GEMM: y = attn_out(144x192) @ proj^T + bias  -> f16 [w][t][C]
__global__ void proj_kernel(const _Float16* __restrict__ ain, const _Float16* __restrict__ wp,
                            const float* __restrict__ bp, _Float16* __restrict__ y) {
  extern __shared__ __align__(16) char smem[];
  _Float16* As = (_Float16*)smem;
  int w = blockIdx.x, tid = threadIdx.x;
  const uint4* srcv = (const uint4*)(ain + (long)w * LWIN * CDIM);
  for (int i = tid; i < LWIN * CDIM / 8; i += 256) cp16_g2l((uint4*)As + i, srcv + i);
  stage_wait();
  int wave = tid >> 5, lane = tid & 31;
  for (int nt = wave; nt < 12; nt += 8) {
    v16h bf[6];
#pragma unroll
    for (int kk = 0; kk < 6; ++kk) bf[kk] = load_frag_b(wp, CDIM, nt * 16, kk * 32, lane);
    int n = nt * 16 + (lane & 15);
    float bias = bp[n];
    for (int mt = 0; mt < 9; ++mt) {
      v8f acc = {};
#pragma unroll
      for (int kk = 0; kk < 6; ++kk) {
        v16h a = load_frag_a(As, CDIM, mt * 16, kk * 32, lane);
        acc = WMMA_F16(a, bf[kk], acc);
      }
      int mb = mt * 16 + 8 * (lane >> 4);
      _Float16* op = y + (long)w * LWIN * CDIM + n;
#pragma unroll
      for (int j = 0; j < 8; ++j) op[(long)(mb + j) * CDIM] = (_Float16)(acc[j] + bias);
    }
  }
}

// x1 = shortcut + window_reverse(y), stored f16 for the MLP GEMM A-operand
__global__ void resid_kernel(const float* __restrict__ x, const _Float16* __restrict__ y,
                             _Float16* __restrict__ x1h) {
  long i = (long)blockIdx.x * blockDim.x + threadIdx.x;
  if (i >= (long)TOK * CDIM) return;
  int n = (int)(i / CDIM), c = (int)(i % CDIM);
  int w, t; tok_to_win(n, w, t);
  x1h[i] = (_Float16)(x[i] + (float)y[((long)w * LWIN + t) * CDIM + c]);
}

// fc1 GEMM + exact GELU: h1 = gelu(x1 @ fc1^T + b)  (K=192, N=768)
__global__ void fc1_kernel(const _Float16* __restrict__ x1h, const _Float16* __restrict__ w1,
                           const float* __restrict__ b1, _Float16* __restrict__ h1) {
  __shared__ __align__(16) _Float16 As[16 * CDIM];
  int m0 = blockIdx.x * 16, tid = threadIdx.x;
  const uint4* srcv = (const uint4*)(x1h + (long)m0 * CDIM);
  for (int i = tid; i < 16 * CDIM / 8; i += 256) cp16_g2l((uint4*)As + i, srcv + i);
  stage_wait();
  int wave = tid >> 5, lane = tid & 31;
  int nt = blockIdx.y * 8 + wave;                       // 0..47
  v8f acc = {};
#pragma unroll
  for (int ks = 0; ks < CDIM; ks += 32) {
    v16h a  = load_frag_a(As, CDIM, 0, ks, lane);
    v16h bf = load_frag_b(w1, CDIM, nt * 16, ks, lane);
    acc = WMMA_F16(a, bf, acc);
  }
  int n = nt * 16 + (lane & 15);
  float bias = b1[n];
  int mb = 8 * (lane >> 4);
  _Float16* op = h1 + (long)m0 * CH + n;
#pragma unroll
  for (int j = 0; j < 8; ++j) {
    float u = acc[j] + bias;
    float ge = 0.5f * u * (1.f + erff(u * 0.7071067811865476f));
    op[(long)(mb + j) * CH] = (_Float16)ge;
  }
}

// fc2 GEMM (K=768,N=192) fused with final post-norm residual: out = x1 + LN(h2)*g + b
__global__ void fc2_final_kernel(const _Float16* __restrict__ h1, const _Float16* __restrict__ w2,
                                 const float* __restrict__ b2, const float* __restrict__ x,
                                 const _Float16* __restrict__ y, const float* __restrict__ g2,
                                 const float* __restrict__ bb2, float* __restrict__ out) {
  __shared__ __align__(16) _Float16 As[16 * CH];        // 24 KB
  __shared__ float H2[16 * CDIM];                       // 12 KB
  int m0 = blockIdx.x * 16, tid = threadIdx.x;
  const uint4* srcv = (const uint4*)(h1 + (long)m0 * CH);
  for (int i = tid; i < 16 * CH / 8; i += 256) cp16_g2l((uint4*)As + i, srcv + i);
  stage_wait();
  int wave = tid >> 5, lane = tid & 31;
  for (int nt = wave; nt < 12; nt += 8) {
    v8f acc = {};
#pragma unroll 4
    for (int ks = 0; ks < CH; ks += 32) {
      v16h a  = load_frag_a(As, CH, 0, ks, lane);
      v16h bf = load_frag_b(w2, CH, nt * 16, ks, lane);
      acc = WMMA_F16(a, bf, acc);
    }
    int n = nt * 16 + (lane & 15);
    float bias = b2[n];
    int mb = 8 * (lane >> 4);
#pragma unroll
    for (int j = 0; j < 8; ++j) H2[(mb + j) * CDIM + n] = acc[j] + bias;
  }
  __syncthreads();
  // per-token LayerNorm of h2, add recomputed x1 = x + y
  for (int tt = wave; tt < 16; tt += 8) {
    int n = m0 + tt;
    int w, t; tok_to_win(n, w, t);
    const float* hr = H2 + tt * CDIM;
    float vals[6];
#pragma unroll
    for (int i = 0; i < 6; ++i) vals[i] = hr[lane + 32 * i];
    float s = 0.f;
#pragma unroll
    for (int i = 0; i < 6; ++i) s += vals[i];
    float mean = wave_sum(s) * (1.f / CDIM);
    float vs = 0.f;
#pragma unroll
    for (int i = 0; i < 6; ++i) { float d = vals[i] - mean; vs += d * d; }
    float inv = rsqrtf(wave_sum(vs) * (1.f / CDIM) + 1e-5f);
    const _Float16* yp = y + ((long)w * LWIN + t) * CDIM;
    const float* xp = x + (long)n * CDIM;
    float* op = out + (long)n * CDIM;
#pragma unroll
    for (int i = 0; i < 6; ++i) {
      int c = lane + 32 * i;
      float x1v = xp[c] + (float)yp[c];
      op[c] = x1v + (vals[i] - mean) * inv * g2[c] + bb2[c];
    }
  }
}

// ---------------- host launcher ----------------
extern "C" void kernel_launch(void* const* d_in, const int* in_sizes, int n_in,
                              void* d_out, int out_size, void* d_ws, size_t ws_size,
                              hipStream_t stream) {
  (void)in_sizes; (void)n_in; (void)out_size; (void)ws_size;
  const float* x     = (const float*)d_in[0];
  const float* n1g   = (const float*)d_in[1];
  const float* n1b   = (const float*)d_in[2];
  const float* qkv_w = (const float*)d_in[3];
  const float* qkv_b = (const float*)d_in[4];
  const float* prj_w = (const float*)d_in[5];
  const float* prj_b = (const float*)d_in[6];
  const float* n2g   = (const float*)d_in[7];
  const float* n2b   = (const float*)d_in[8];
  const float* fc1_w = (const float*)d_in[9];
  const float* fc1_b = (const float*)d_in[10];
  const float* fc2_w = (const float*)d_in[11];
  const float* fc2_b = (const float*)d_in[12];
  float* out = (float*)d_out;

  char* ws = (char*)d_ws;
  size_t off = 0;
  auto alloc = [&](size_t bytes) { size_t o = off; off = (off + bytes + 255) & ~(size_t)255; return o; };
  const long WE = (long)NWIN * LWIN * CDIM;             // 26,542,080 elems
  _Float16* wqkv_h = (_Float16*)(ws + alloc((size_t)3 * CDIM * CDIM * 2));
  _Float16* wprj_h = (_Float16*)(ws + alloc((size_t)CDIM * CDIM * 2));
  _Float16* wfc1_h = (_Float16*)(ws + alloc((size_t)CH * CDIM * 2));
  _Float16* wfc2_h = (_Float16*)(ws + alloc((size_t)CDIM * CH * 2));
  _Float16* xw  = (_Float16*)(ws + alloc((size_t)WE * 2));   // reused as attn_out
  _Float16* qb  = (_Float16*)(ws + alloc((size_t)WE * 2));   // reused as proj output y
  _Float16* kb  = (_Float16*)(ws + alloc((size_t)WE * 2));
  _Float16* vb  = (_Float16*)(ws + alloc((size_t)WE * 2));
  _Float16* x1h = (_Float16*)(ws + alloc((size_t)TOK * CDIM * 2));
  _Float16* h1  = (_Float16*)(ws + alloc((size_t)TOK * CH * 2));
  _Float16* attn_out = xw;
  _Float16* ybuf = qb;

  const int SM_QKV  = LWIN * CDIM * 2;                  // 55,296 B
  const int SM_ATTN = 166912;                           // q+k+vT+S(f32)+P
  (void)hipFuncSetAttribute((const void*)qkv_kernel,  hipFuncAttributeMaxDynamicSharedMemorySize, SM_QKV);
  (void)hipFuncSetAttribute((const void*)proj_kernel, hipFuncAttributeMaxDynamicSharedMemorySize, SM_QKV);
  (void)hipFuncSetAttribute((const void*)attn_kernel, hipFuncAttributeMaxDynamicSharedMemorySize, SM_ATTN);

  // 1) weight f32 -> f16 (row-major [N,K] matches WMMA B-fragment layout directly)
  cvt_f16_kernel<<<(3 * CDIM * CDIM + 255) / 256, 256, 0, stream>>>(qkv_w, wqkv_h, 3 * CDIM * CDIM);
  cvt_f16_kernel<<<(CDIM * CDIM + 255) / 256, 256, 0, stream>>>(prj_w, wprj_h, CDIM * CDIM);
  cvt_f16_kernel<<<(CH * CDIM + 255) / 256, 256, 0, stream>>>(fc1_w, wfc1_h, CH * CDIM);
  cvt_f16_kernel<<<(CDIM * CH + 255) / 256, 256, 0, stream>>>(fc2_w, wfc2_h, CDIM * CH);

  // 2) roll/pad/window-gather + LayerNorm (one wave per token, 8 waves/block)
  ln_window_kernel<<<NWIN * LWIN / 8, 256, 0, stream>>>(x, n1g, n1b, xw);

  // 3) QKV GEMM per window
  qkv_kernel<<<NWIN, 256, SM_QKV, stream>>>(xw, wqkv_h, qkv_b, qb, kb, vb);

  // 4) attention per (window, head)
  attn_kernel<<<NWIN * NHEAD, 256, SM_ATTN, stream>>>(qb, kb, vb, attn_out);

  // 5) proj GEMM per window (y aliases the q buffer — q already consumed)
  proj_kernel<<<NWIN, 256, SM_QKV, stream>>>(attn_out, wprj_h, prj_b, ybuf);

  // 6) window reverse + shortcut -> x1 (f16)
  resid_kernel<<<(int)(((long)TOK * CDIM + 255) / 256), 256, 0, stream>>>(x, ybuf, x1h);

  // 7) fc1 + exact GELU
  fc1_kernel<<<dim3(TOK / 16, CH / (16 * 8)), 256, 0, stream>>>(x1h, wfc1_h, fc1_b, h1);

  // 8) fc2 + post-norm LayerNorm + residual -> out (f32)
  fc2_final_kernel<<<TOK / 16, 256, 0, stream>>>(h1, wfc2_h, fc2_b, x, ybuf, n2g, n2b, out);
}
